// MultiHeadSelfAttentionEdges_27127013441543
// MI455X (gfx1250) — matrix operand order repaired and checked by reading, not compiled, not run on hardware
//
#include <hip/hip_runtime.h>
#include <math.h>

// ---- problem constants (B=1, M=N=192, D=384, H=12, hd=32) ----
#define NDIM 192
#define DDIM 384
#define HH   12
#define HDIM 32
#define STOK (NDIM * NDIM)          // 36864 tokens
#define SD   ((size_t)STOK * DDIM)  // 14,155,776 elements

typedef unsigned short u16;
typedef __bf16 bf16x16 __attribute__((ext_vector_type(16)));
typedef float  f32x8   __attribute__((ext_vector_type(8)));
typedef unsigned int u32x4 __attribute__((ext_vector_type(4)));
typedef int i32x4 __attribute__((ext_vector_type(4)));
typedef int i32x8 __attribute__((ext_vector_type(8)));

union FragU { bf16x16 v; uint4 q[2]; };

__device__ __forceinline__ u16 f2bf(float f) {
  unsigned int x = __builtin_bit_cast(unsigned int, f);
  x += 0x7fffu + ((x >> 16) & 1u);          // round-to-nearest-even
  return (u16)(x >> 16);
}
__device__ __forceinline__ float b2f(u16 u) {
  unsigned int x = ((unsigned int)u) << 16;
  return __builtin_bit_cast(float, x);
}
__device__ __forceinline__ f32x8 zero8() {
  f32x8 z; for (int e = 0; e < 8; e++) z[e] = 0.0f; return z;
}

// ----------------------------------------------------------------------------
// TDM: load a 2D tile (tile_h rows x tile_w bf16 elems, row stride = stride
// elems) from global memory into LDS at byte offset lds_off.
// D# layout per CDNA5 ISA (2D, groups 2/3 zero).
// This toolchain exposes the 6-arg builtin:
//   (uint32x4 g0, int32x8 g1, int32x4 g2, int32x4 g3, int32x8 g4, i32 cpol)
// ----------------------------------------------------------------------------
__device__ __forceinline__ void tdm_load_2d_bf16(unsigned lds_off,
                                                 const void* gptr,
                                                 unsigned tile_w, unsigned tile_h,
                                                 unsigned stride_elems)
{
  unsigned long long ga = (unsigned long long)gptr;
  const unsigned dim0 = 1u << 20, dim1 = 1u << 20;  // large: no OOB clipping
  u32x4 g0;
  g0[0] = 1u;                                        // count=1, user mode
  g0[1] = lds_off;                                   // lds_addr [63:32]
  g0[2] = (unsigned)ga;                              // global_addr lo
  g0[3] = ((unsigned)(ga >> 32) & 0x01FFFFFFu)       // global_addr[56:32]
          | 0x80000000u;                             // type=2 ("image")
  i32x8 g1;
  g1[0] = 0x00010000;                                // data_size=1 -> 2 bytes
  g1[1] = (int)((dim0 & 0xFFFFu) << 16);             // tensor_dim0 lo16
  g1[2] = (int)(((dim0 >> 16) & 0xFFFFu) | ((dim1 & 0xFFFFu) << 16));
  g1[3] = (int)(((dim1 >> 16) & 0xFFFFu) | (tile_w << 16));   // tile_dim0
  g1[4] = (int)(tile_h);                             // tile_dim1 (tile_dim2=0)
  g1[5] = (int)stride_elems;                         // tensor_dim0_stride lo32
  g1[6] = 0;                                         // stride hi16 | dim1_stride lo16
  g1[7] = 0;                                         // tensor_dim1_stride hi32
  i32x4 gz; gz[0] = 0; gz[1] = 0; gz[2] = 0; gz[3] = 0;
  i32x8 gz8;
  for (int e = 0; e < 8; e++) gz8[e] = 0;
  __builtin_amdgcn_tensor_load_to_lds(g0, g1, gz, gz, gz8, 0);
}

// ============================================================================
// elementwise f32 -> bf16 (vectorized x4)
// ============================================================================
__global__ void cvt_kernel(const float* __restrict__ in, u16* __restrict__ out,
                           size_t n4)
{
  size_t i = (size_t)blockIdx.x * blockDim.x + threadIdx.x;
  if (i >= n4) return;
  float4 v = ((const float4*)in)[i];
  u16 t[4] = { f2bf(v.x), f2bf(v.y), f2bf(v.z), f2bf(v.w) };
  *(uint2*)&out[i * 4] = *(const uint2*)t;
}

// ============================================================================
// weight convert + transpose: Wt[n*K + k] = bf16(W[k*N + n])
// ============================================================================
__global__ void cvtT_kernel(const float* __restrict__ W, u16* __restrict__ Wt,
                            int K, int N)
{
  int idx = blockIdx.x * blockDim.x + threadIdx.x;
  if (idx >= K * N) return;
  int n = idx / K, k = idx % K;
  Wt[idx] = f2bf(W[(size_t)k * N + n]);
}

// ============================================================================
// GEMM: C[M x N] = A[M x K] (bf16 row-major) * Bt[N x K]^T (bf16, pre-transposed)
// Block: 128 threads (4 waves), tile 64x64; tiles staged to LDS by the TDM.
// epi: 0 -> bf16 store; 1 -> bf16 sigmoid(v + bias_vec); 2 -> f32 (v + bias_vec)
// ============================================================================
__global__ __launch_bounds__(128)
void gemm_wmma_kernel(const u16* __restrict__ A, const u16* __restrict__ Bt,
                      void* __restrict__ Cptr, int epi,
                      const float* __restrict__ bias_vec,
                      int Kdim, int Ncols)
{
  __shared__ u16 As[64 * 32];   // As[r*32 + k]
  __shared__ u16 Bs[64 * 32];   // Bs[n*32 + k]

  const int tid  = threadIdx.x;
  const int wid  = tid >> 5;
  const int lane = tid & 31;
  const int m    = lane & 15;
  const int half = lane >> 4;
  const int row0 = blockIdx.x * 64;
  const int col0 = blockIdx.y * 64;

  const unsigned asOff = (unsigned)(size_t)(void*)As;
  const unsigned bsOff = (unsigned)(size_t)(void*)Bs;

  f32x8 acc[4];
  for (int i = 0; i < 4; i++) acc[i] = zero8();

  for (int kk = 0; kk < Kdim; kk += 32) {
    if (wid == 0) {
      tdm_load_2d_bf16(asOff, A  + (size_t)row0 * Kdim + kk, 32, 64, Kdim);
      tdm_load_2d_bf16(bsOff, Bt + (size_t)col0 * Kdim + kk, 32, 64, Kdim);
      __builtin_amdgcn_s_wait_tensorcnt(0);
    }
    __syncthreads();

    // A fragment (16x32 bf16, ISA 7.12.2): elems 0-7 -> K=half*8+0..7,
    // elems 8-15 -> K=16+half*8+0..7
    FragU a;
    a.q[0] = *(const uint4*)&As[(wid * 16 + m) * 32 + half * 8];
    a.q[1] = *(const uint4*)&As[(wid * 16 + m) * 32 + 16 + half * 8];
    #pragma unroll
    for (int ct = 0; ct < 4; ct++) {
      // B fragment (32x16): lane n holds column n, elems 0-7 -> K=half*8+0..7, elems 8-15 -> K=16+half*8+0..7
      FragU b;
      b.q[0] = *(const uint4*)&Bs[(ct * 16 + m) * 32 + half * 8];
      b.q[1] = *(const uint4*)&Bs[(ct * 16 + m) * 32 + 16 + half * 8];
      acc[ct] = __builtin_amdgcn_wmma_f32_16x16x32_bf16(
          false, a.v, false, b.v, (short)0, acc[ct], false, false);
    }
    __syncthreads();
  }

  // epilogue: D layout -> VGPR r holds M = r + 8*half, N = lane&15
  #pragma unroll
  for (int ct = 0; ct < 4; ct++) {
    #pragma unroll
    for (int r = 0; r < 8; r++) {
      int row = row0 + wid * 16 + half * 8 + r;
      int col = col0 + ct * 16 + m;
      float v  = acc[ct][r];
      float bv = bias_vec ? bias_vec[col] : 0.0f;
      if (epi == 2) {
        ((float*)Cptr)[(size_t)row * Ncols + col] = v + bv;
      } else if (epi == 1) {
        float g = 1.0f / (1.0f + __expf(-(v + bv)));
        ((u16*)Cptr)[(size_t)row * Ncols + col] = f2bf(g);
      } else {
        ((u16*)Cptr)[(size_t)row * Ncols + col] = f2bf(v);
      }
    }
  }
}

// ============================================================================
// bias = E @ Wb : (36864 x 384) @ (384 x 12); writes h-major + transposed copy:
//   biasA [h*STOK + i*192 + j] and biasAT[h*STOK + j*192 + i]
// ============================================================================
__global__ void bias_proj_kernel(const float* __restrict__ E,
                                 const float* __restrict__ Wb,
                                 float* __restrict__ biasA,
                                 float* __restrict__ biasAT)
{
  int idx = blockIdx.x * blockDim.x + threadIdx.x;
  if (idx >= STOK * HH) return;
  int s = idx / HH, h = idx % HH;
  const float* e = E + (size_t)s * DDIM;
  float sum = 0.0f;
  for (int d = 0; d < DDIM; d++) sum += e[d] * Wb[d * HH + h];
  int i = s / NDIM, j = s % NDIM;
  biasA [(size_t)h * STOK + i * NDIM + j] = sum;
  biasAT[(size_t)h * STOK + j * NDIM + i] = sum;
}

// ============================================================================
// Attention: one block per (mode, head, pivot). 4 waves, 128 threads.
// LDS: Qs 192x32 | Ks 192x32 | Vt 32x192 (bf16) | Pb 192x192 bf16
//      maskAdd[192] f32 | Sc 192x192 f32   => 258,816 B < 320 KB WGP LDS.
// mode 0 (row, pivot=i): tok(x)=i*192+x ; bias=biasA [h][x*192+y]; mask[i*192+y]
// mode 1 (col, pivot=j): tok(x)=x*192+j ; bias=biasAT[h][x*192+y]; mask[y*192+j]
// ============================================================================
__global__ __launch_bounds__(128)
void attn_kernel(const u16* __restrict__ Qb, const u16* __restrict__ Kb,
                 const u16* __restrict__ Vb, const float* __restrict__ biasA,
                 const float* __restrict__ biasAT,
                 const unsigned char* __restrict__ mask,
                 u16* __restrict__ Orow, u16* __restrict__ Ocol)
{
  extern __shared__ unsigned char smem[];
  u16*   Qs      = (u16*)smem;                  // 192*32
  u16*   Ks      = Qs + 192 * 32;               // 192*32
  u16*   Vt      = Ks + 192 * 32;               // 32*192  Vt[d*192 + y]
  u16*   Pb      = Vt + 32 * 192;               // 192*192
  float* maskAdd = (float*)(Pb + 192 * 192);    // 192
  float* Sc      = maskAdd + 192;               // 192*192

  const int tid  = threadIdx.x;
  const int wid  = tid >> 5;
  const int lane = tid & 31;
  const int m    = lane & 15;
  const int half = lane >> 4;

  const int bid  = blockIdx.x;
  const int mode = bid / (HH * NDIM);
  const int rem  = bid % (HH * NDIM);
  const int h    = rem / NDIM;
  const int p    = rem % NDIM;

  auto sTok = [&](int x) { return (mode == 0) ? p * NDIM + x : x * NDIM + p; };

  // ---- stage Q, K, V(^T) for this (mode, h, p) ----
  for (int t = tid; t < NDIM * 4; t += 128) {
    int x = t >> 2, c = (t & 3) * 8;
    size_t base = (size_t)sTok(x) * DDIM + h * HDIM + c;
    *(uint4*)&Qs[x * 32 + c] = *(const uint4*)&Qb[base];
    *(uint4*)&Ks[x * 32 + c] = *(const uint4*)&Kb[base];
    uint4 vv = *(const uint4*)&Vb[base];
    const u16* ve = (const u16*)&vv;
    #pragma unroll
    for (int e = 0; e < 8; e++) Vt[(c + e) * NDIM + x] = ve[e];
  }
  for (int t = tid; t < NDIM; t += 128) {
    unsigned char mv = (mode == 0) ? mask[p * NDIM + t] : mask[t * NDIM + p];
    maskAdd[t] = mv ? -1e30f : 0.0f;
  }
  __syncthreads();

  const float scale = 0.17677669529663688f;  // 1/sqrt(32)
  const float* bsel = (mode == 0) ? (biasA + (size_t)h * STOK)
                                  : (biasAT + (size_t)h * STOK);

  // ---- scores: S[x][y] = scale*Q.K + bias + mask  (144 WMMA tiles / 4 waves)
  for (int tile = wid; tile < 144; tile += 4) {
    int xt = tile / 12, yt = tile % 12;
    FragU a, b;
    a.q[0] = *(const uint4*)&Qs[(xt * 16 + m) * 32 + half * 8];
    a.q[1] = *(const uint4*)&Qs[(xt * 16 + m) * 32 + 16 + half * 8];
    b.q[0] = *(const uint4*)&Ks[(yt * 16 + m) * 32 + half * 8];
    b.q[1] = *(const uint4*)&Ks[(yt * 16 + m) * 32 + 16 + half * 8];
    f32x8 acc = zero8();
    acc = __builtin_amdgcn_wmma_f32_16x16x32_bf16(
        false, a.v, false, b.v, (short)0, acc, false, false);
    #pragma unroll
    for (int r = 0; r < 8; r++) {
      int x = xt * 16 + half * 8 + r;
      int y = yt * 16 + m;
      Sc[x * NDIM + y] = acc[r] * scale + bsel[x * NDIM + y] + maskAdd[y];
    }
  }
  __syncthreads();

  // ---- softmax over y for each row x ----
  for (int x = tid; x < NDIM; x += 128) {
    float* row = Sc + x * NDIM;
    float mx = -3.4e38f;
    for (int y = 0; y < NDIM; y++) mx = fmaxf(mx, row[y]);
    float sum = 0.0f;
    for (int y = 0; y < NDIM; y++) { float e = __expf(row[y] - mx); row[y] = e; sum += e; }
    float inv = 1.0f / sum;
    for (int y = 0; y < NDIM; y++) Pb[x * NDIM + y] = f2bf(row[y] * inv);
  }
  __syncthreads();

  // ---- O = P @ V : 12 x-tiles * 2 d-tiles, K-loop 192/32 = 6 WMMAs each ----
  u16* Obuf = (mode == 0) ? Orow : Ocol;
  for (int tile = wid; tile < 24; tile += 4) {
    int xt = tile >> 1, dt = tile & 1;
    f32x8 acc = zero8();
    for (int kk = 0; kk < NDIM; kk += 32) {
      FragU a, b;
      a.q[0] = *(const uint4*)&Pb[(xt * 16 + m) * NDIM + kk + half * 8];
      a.q[1] = *(const uint4*)&Pb[(xt * 16 + m) * NDIM + kk + 16 + half * 8];
      b.q[0] = *(const uint4*)&Vt[(dt * 16 + m) * NDIM + kk + half * 8];
      b.q[1] = *(const uint4*)&Vt[(dt * 16 + m) * NDIM + kk + 16 + half * 8];
      acc = __builtin_amdgcn_wmma_f32_16x16x32_bf16(
          false, a.v, false, b.v, (short)0, acc, false, false);
    }
    #pragma unroll
    for (int r = 0; r < 8; r++) {
      int x = xt * 16 + half * 8 + r;
      int d = dt * 16 + m;
      Obuf[(size_t)sTok(x) * DDIM + h * HDIM + d] = f2bf(acc[r]);
    }
  }
}

// ============================================================================
// X = (Orow + Ocol) * gate   (all bf16, f32 math)
// ============================================================================
__global__ void combine_kernel(const u16* __restrict__ Orow,
                               const u16* __restrict__ Ocol,
                               const u16* __restrict__ Gate,
                               u16* __restrict__ Xb, size_t n)
{
  size_t i = (size_t)blockIdx.x * blockDim.x + threadIdx.x;
  if (i >= n) return;
  float v = (b2f(Orow[i]) + b2f(Ocol[i])) * b2f(Gate[i]);
  Xb[i] = f2bf(v);
}

// ============================================================================
extern "C" void kernel_launch(void* const* d_in, const int* in_sizes, int n_in,
                              void* d_out, int out_size, void* d_ws, size_t ws_size,
                              hipStream_t stream)
{
  (void)in_sizes; (void)n_in; (void)out_size; (void)ws_size;
  const float* E    = (const float*)d_in[0];
  const float* Wq   = (const float*)d_in[1];
  const float* Wk   = (const float*)d_in[2];
  const float* Wv   = (const float*)d_in[3];
  const float* Wo   = (const float*)d_in[4];
  const float* bo   = (const float*)d_in[5];
  const float* Wg   = (const float*)d_in[6];
  const float* bg   = (const float*)d_in[7];
  const float* Wb   = (const float*)d_in[8];
  const unsigned char* mask = (const unsigned char*)d_in[9];
  float* out = (float*)d_out;

  // workspace partition (256B aligned)
  char* ws = (char*)d_ws;
  auto alloc = [&](size_t bytes) -> char* {
    char* r = ws; ws += (bytes + 255) & ~(size_t)255; return r;
  };
  u16* Ebf     = (u16*)alloc(SD * 2);
  u16* Qbf     = (u16*)alloc(SD * 2);
  u16* Kbf     = (u16*)alloc(SD * 2);
  u16* Vbf     = (u16*)alloc(SD * 2);
  u16* Gatebf  = (u16*)alloc(SD * 2);
  u16* OrowB   = (u16*)alloc(SD * 2);
  u16* OcolB   = (u16*)alloc(SD * 2);
  u16* Xbf     = (u16*)alloc(SD * 2);
  u16* WqT     = (u16*)alloc((size_t)DDIM * DDIM * 2);
  u16* WkT     = (u16*)alloc((size_t)DDIM * DDIM * 2);
  u16* WvT     = (u16*)alloc((size_t)DDIM * DDIM * 2);
  u16* WgT     = (u16*)alloc((size_t)DDIM * DDIM * 2);
  u16* WoT     = (u16*)alloc((size_t)DDIM * DDIM * 2);
  float* biasA  = (float*)alloc((size_t)STOK * HH * 4);
  float* biasAT = (float*)alloc((size_t)STOK * HH * 4);

  // one-time conversions (E once, each weight once)
  cvt_kernel<<<(unsigned)((SD / 4 + 255) / 256), 256, 0, stream>>>(E, Ebf, SD / 4);
  int wn = DDIM * DDIM, wblk = (wn + 255) / 256;
  cvtT_kernel<<<wblk, 256, 0, stream>>>(Wq, WqT, DDIM, DDIM);
  cvtT_kernel<<<wblk, 256, 0, stream>>>(Wk, WkT, DDIM, DDIM);
  cvtT_kernel<<<wblk, 256, 0, stream>>>(Wv, WvT, DDIM, DDIM);
  cvtT_kernel<<<wblk, 256, 0, stream>>>(Wg, WgT, DDIM, DDIM);
  cvtT_kernel<<<wblk, 256, 0, stream>>>(Wo, WoT, DDIM, DDIM);

  dim3 gGrid(STOK / 64, DDIM / 64);  // (576, 6)

  // projections (TDM-staged WMMA GEMMs)
  gemm_wmma_kernel<<<gGrid, 128, 0, stream>>>(Ebf, WqT, Qbf,    0, nullptr, DDIM, DDIM);
  gemm_wmma_kernel<<<gGrid, 128, 0, stream>>>(Ebf, WkT, Kbf,    0, nullptr, DDIM, DDIM);
  gemm_wmma_kernel<<<gGrid, 128, 0, stream>>>(Ebf, WvT, Vbf,    0, nullptr, DDIM, DDIM);
  gemm_wmma_kernel<<<gGrid, 128, 0, stream>>>(Ebf, WgT, Gatebf, 1, bg,      DDIM, DDIM);
  bias_proj_kernel<<<(STOK * HH + 255) / 256, 256, 0, stream>>>(E, Wb, biasA, biasAT);

  // attention (row + col modes)
  size_t smem = (size_t)(192 * 32 * 3 + 192 * 192) * 2 + (size_t)(192 + 192 * 192) * 4;
  hipError_t rc = hipFuncSetAttribute(reinterpret_cast<const void*>(&attn_kernel),
                                      hipFuncAttributeMaxDynamicSharedMemorySize,
                                      (int)smem);
  (void)rc;
  attn_kernel<<<2 * HH * NDIM, 128, smem, stream>>>(Qbf, Kbf, Vbf, biasA, biasAT,
                                                    mask, OrowB, OcolB);

  // gate + output projection
  combine_kernel<<<(unsigned)((SD + 255) / 256), 256, 0, stream>>>(OrowB, OcolB,
                                                                   Gatebf, Xbf, SD);
  gemm_wmma_kernel<<<gGrid, 128, 0, stream>>>(Xbf, WoT, out, 2, bo, DDIM, DDIM);
}
